// TopologicalFeatureExtractor_7421703488067
// MI455X (gfx1250) — matrix operand order repaired and checked by reading, not compiled, not run on hardware
//
#include <hip/hip_runtime.h>

#define BATCH 8
#define SEQ   2048
#define EMB   768
#define TOPO  256
#define KNBR  32
#define ROWS  (BATCH*SEQ)   // 16384

typedef _Float16 f16;
typedef __attribute__((ext_vector_type(16))) _Float16 v16h;
typedef __attribute__((ext_vector_type(8)))  float    v8f;

union HU { v16h v; unsigned u[8]; };

#define LPITCH 40   // f16 units (20 dwords, 80B): conflict-free frags + 16B-aligned b128 chunks

// ---- CDNA5 async global->LDS copy (ASYNCcnt tracked), per-lane b128 ----
static __device__ inline void async_load_b128(unsigned ldsOff, const void* gaddr) {
  asm volatile("global_load_async_to_lds_b128 %0, %1, off"
               :: "v"(ldsOff), "v"((unsigned long long)(uintptr_t)gaddr)
               : "memory");
}
static __device__ inline void wait_async0() { asm volatile("s_wait_asynccnt 0x0" ::: "memory"); }
static __device__ inline void wait_async2() { asm volatile("s_wait_asynccnt 0x2" ::: "memory"); }
static __device__ inline void wait_async3() { asm volatile("s_wait_asynccnt 0x3" ::: "memory"); }
static __device__ inline unsigned lds_off(const void* p) {
  return (unsigned)(uintptr_t)p;     // low 32 bits of flat LDS address = LDS byte offset
}

// A-matrix fragment (16xK=32, f16): lane m=lane&15, half=lane>>4
// element e -> K = e + 8*half (e<8) ; K = e+8 + 8*half (e>=8)
static __device__ inline v16h frag_a(const f16* As, int mBase, int lane) {
  const int m = mBase + (lane & 15);
  const int h = (lane >> 4) & 1;
  const unsigned* p = (const unsigned*)(As + m * LPITCH);
  HU r;
#pragma unroll
  for (int j = 0; j < 4; ++j) r.u[j]     = p[4*h + j];
#pragma unroll
  for (int j = 0; j < 4; ++j) r.u[4 + j] = p[8 + 4*h + j];
  return r.v;
}

// B-matrix fragment (K=32 x 16, f16) from LDS staged transposed BsT[n][k]:
// lane n=lane&15, half=lane>>4 ; element e -> K = e + 8*half (e<8) ; K = e+8 + 8*half (e>=8)
static __device__ inline v16h frag_b(const f16* BsT, int nBase, int lane) {
  const int n = nBase + (lane & 15);
  const int h = (lane >> 4) & 1;
  const unsigned* p = (const unsigned*)(BsT + n * LPITCH);
  HU r;
#pragma unroll
  for (int j = 0; j < 4; ++j) r.u[j]     = p[4*h + j];
#pragma unroll
  for (int j = 0; j < 4; ++j) r.u[4 + j] = p[8 + 4*h + j];
  return r.v;
}

#define WMMA_F16(a,b,c) __builtin_amdgcn_wmma_f32_16x16x32_f16(false,(a),false,(b),(short)0,(c),false,false)

// ---------------------------------------------------------------------------
// f16 WMMA GEMM: C[M,N] = A[M,Kd]*B[Kd,N] + bias, compile-time RELU / out type.
// Block tile 128x64, 8 waves, 32x32 per wave. M%128==0, N%64==0, Kd%32==0.
// Double-buffered LDS: async A copies for step k+1 overlap WMMA of step k
// (each wave issues exactly 2 asyncs/step; s_wait_asynccnt 2 retires step k).
// ---------------------------------------------------------------------------
template<bool RELU, bool F16OUT>
__global__ __launch_bounds__(256)
void wmma_gemm_f16(const f16* __restrict__ A, const f16* __restrict__ Bw,
                   const float* __restrict__ bias, void* __restrict__ Cout,
                   int N, int Kd)
{
  __shared__ f16 As[2][128 * LPITCH];
  __shared__ f16 Bs[2][64 * LPITCH];
  const int t    = threadIdx.x;
  const int lane = t & 31;
  const int w    = t >> 5;
  const int wm   = w & 3;     // 4 waves along M (32 rows each)
  const int wn   = w >> 2;    // 2 waves along N (32 cols each)
  const int mBase = blockIdx.y * 128;
  const int nBase = blockIdx.x * 64;
  const int kd2   = Kd >> 1;
  const int nsteps = Kd >> 5;

  v8f acc[2][2];
#pragma unroll
  for (int i = 0; i < 2; ++i)
#pragma unroll
    for (int j = 0; j < 2; ++j)
#pragma unroll
      for (int v = 0; v < 8; ++v) acc[i][j][v] = 0.f;

  auto stage = [&](int ks, int buf) {
    const int k0 = ks << 5;
    unsigned* AsU = (unsigned*)As[buf];
    const unsigned* Ag = (const unsigned*)(A + (size_t)mBase * Kd + k0);
#pragma unroll
    for (int i = 0; i < 2; ++i) {              // 512 b128 chunks, 2/thread
      int idx = t + 256 * i;
      int row = idx >> 2, q = idx & 3;
      async_load_b128(lds_off(&AsU[row * 20 + q * 4]),
                      (const void*)(Ag + (size_t)row * kd2 + q * 4));
    }
    f16* Bsb = Bs[buf];                        // transpose -> VGPR path
#pragma unroll
    for (int i = 0; i < 4; ++i) {
      int idx = t + 256 * i;                   // K-pairs
      int kc = idx >> 6, n = idx & 63;
      f16 lo = Bw[(size_t)(k0 + 2*kc)     * N + nBase + n];
      f16 hi = Bw[(size_t)(k0 + 2*kc + 1) * N + nBase + n];
      Bsb[n * LPITCH + 2*kc]     = lo;
      Bsb[n * LPITCH + 2*kc + 1] = hi;
    }
  };

  stage(0, 0);
  for (int ks = 0; ks < nsteps; ++ks) {
    const int cur = ks & 1;
    if (ks + 1 < nsteps) { stage(ks + 1, cur ^ 1); wait_async2(); }
    else                 { wait_async0(); }
    __syncthreads();
    v16h a0 = frag_a(As[cur], wm*32,      lane);
    v16h a1 = frag_a(As[cur], wm*32 + 16, lane);
    v16h b0 = frag_b(Bs[cur], wn*32,      lane);
    v16h b1 = frag_b(Bs[cur], wn*32 + 16, lane);
    acc[0][0] = WMMA_F16(a0, b0, acc[0][0]);
    acc[0][1] = WMMA_F16(a0, b1, acc[0][1]);
    acc[1][0] = WMMA_F16(a1, b0, acc[1][0]);
    acc[1][1] = WMMA_F16(a1, b1, acc[1][1]);
    __syncthreads();
  }

  float* Cf = (float*)Cout;
  f16*   Ch = (f16*)Cout;
  const int h  = (lane >> 4) & 1;
  const int cl = lane & 15;
#pragma unroll
  for (int mi = 0; mi < 2; ++mi)
#pragma unroll
    for (int ni = 0; ni < 2; ++ni) {
      int col = nBase + wn*32 + ni*16 + cl;
      float bv = bias[col];
#pragma unroll
      for (int v = 0; v < 8; ++v) {
        int row = mBase + wm*32 + mi*16 + v + 8*h;
        float val = acc[mi][ni][v] + bv;
        if (RELU) val = fmaxf(val, 0.f);
        size_t o = (size_t)row * N + col;
        if (F16OUT) Ch[o] = (f16)val;
        else        Cf[o] = val;
      }
    }
}

// ---------------------------------------------------------------------------
// Fused cosine-similarity + top-K. One block per (batch, 32 rows).
// Both tiles are contiguous e_norm rows -> fully async-staged, double-buffered
// (every wave issues exactly 3 asyncs/step: A chunk duplicated via t&127).
// 32 owner lanes keep sorted ascending top-32 (stable like argsort).
// ---------------------------------------------------------------------------
__global__ __launch_bounds__(256)
void sim_topk(const f16* __restrict__ enorm, float* __restrict__ outD,
              float* __restrict__ outI)
{
  __shared__ f16   As[2][32 * LPITCH];
  __shared__ f16   Bs[2][128 * LPITCH];
  __shared__ float distBuf[32 * 129];
  __shared__ float topD[32 * KNBR];
  __shared__ int   topI[32 * KNBR];

  const int t = threadIdx.x, lane = t & 31, w = t >> 5;
  const int wr = w >> 2;    // 0..1 : 16-row slab
  const int wc = w & 3;     // 0..3 : 32-col slab
  const int b = blockIdx.x >> 6;
  const int rBase = (blockIdx.x & 63) * 32;
  const unsigned* En = (const unsigned*)(enorm + (size_t)b * SEQ * EMB); // 384 dw/row
  int myLen = 0;
  const int myRow = rBase + t;           // meaningful for t<32

  for (int chunk = 0; chunk < SEQ / 128; ++chunk) {
    const int cBase = chunk * 128;
    v8f acc[2];
#pragma unroll
    for (int j = 0; j < 2; ++j)
#pragma unroll
      for (int v = 0; v < 8; ++v) acc[j][v] = 0.f;

    auto stageS = [&](int ks, int buf) {
      const int kdw = ks << 4;                 // k0/2 dwords
      unsigned* AsU = (unsigned*)As[buf];
      unsigned* BsU = (unsigned*)Bs[buf];
      {                                        // A: 128 chunks, duplicated x2
        int a = t & 127;
        int row = a >> 2, q = a & 3;
        async_load_b128(lds_off(&AsU[row * 20 + q * 4]),
                        (const void*)(En + (size_t)(rBase + row) * 384 + kdw + q * 4));
      }
#pragma unroll
      for (int i = 0; i < 2; ++i) {            // B^T: 512 chunks, 2/thread
        int idx = t + 256 * i;
        int n = idx >> 2, q = idx & 3;
        async_load_b128(lds_off(&BsU[n * 20 + q * 4]),
                        (const void*)(En + (size_t)(cBase + n) * 384 + kdw + q * 4));
      }
    };

    const int nsteps = EMB / 32;               // 24
    stageS(0, 0);
    for (int ks = 0; ks < nsteps; ++ks) {
      const int cur = ks & 1;
      if (ks + 1 < nsteps) { stageS(ks + 1, cur ^ 1); wait_async3(); }
      else                 { wait_async0(); }
      __syncthreads();
      v16h a0 = frag_a(As[cur], wr * 16, lane);
      v16h b0 = frag_b(Bs[cur], wc * 32,      lane);
      v16h b1 = frag_b(Bs[cur], wc * 32 + 16, lane);
      acc[0] = WMMA_F16(a0, b0, acc[0]);
      acc[1] = WMMA_F16(a0, b1, acc[1]);
      __syncthreads();
    }

    const int h = (lane >> 4) & 1, cl = lane & 15;
#pragma unroll
    for (int ni = 0; ni < 2; ++ni)
#pragma unroll
      for (int v = 0; v < 8; ++v) {
        int r = wr * 16 + v + 8 * h;
        int c = wc * 32 + ni * 16 + cl;
        distBuf[r * 129 + c] = 1.0f - acc[ni][v];
      }
    __syncthreads();

    if (t < 32) {
      const int base = t * KNBR;
      for (int c = 0; c < 128; ++c) {
        int gc = cBase + c;
        if (gc == myRow) continue;                     // diagonal masked (1e9)
        float d = distBuf[t * 129 + c];
        if (myLen == KNBR && d >= topD[base + KNBR - 1]) continue;
        int i = (myLen < KNBR) ? myLen : (KNBR - 1);
        while (i > 0 && topD[base + i - 1] > d) {      // strict > => stable order
          topD[base + i] = topD[base + i - 1];
          topI[base + i] = topI[base + i - 1];
          --i;
        }
        topD[base + i] = d;
        topI[base + i] = gc;
        if (myLen < KNBR) ++myLen;
      }
    }
    __syncthreads();
  }

  if (t < 32) {
    size_t o = ((size_t)b * SEQ + myRow) * KNBR;
    for (int j = 0; j < KNBR; ++j) {
      outD[o + j] = topD[t * KNBR + j];
      outI[o + j] = (float)topI[t * KNBR + j];
    }
  }
}

// ---------------------------------------------------------------------------
// combined = topo + softmax(-dist)-weighted neighbor topo. Block per (b,s).
// ---------------------------------------------------------------------------
__global__ __launch_bounds__(256)
void combine_neighbors(const float* __restrict__ topo,
                       const float* __restrict__ nbrD,
                       const float* __restrict__ nbrI,
                       f16* __restrict__ combH)
{
  __shared__ float wgt[KNBR];
  __shared__ int   idx[KNBR];
  __shared__ float ssum;
  const int row = blockIdx.x;         // b*SEQ + s
  const int b   = row >> 11;
  const int t   = threadIdx.x;
  if (t < KNBR) {
    float d0 = nbrD[(size_t)row * KNBR];               // sorted ascending -> min
    float d  = nbrD[(size_t)row * KNBR + t];
    wgt[t] = expf(d0 - d);
    idx[t] = (int)(nbrI[(size_t)row * KNBR + t] + 0.5f);
  }
  __syncthreads();
  if (t == 0) {
    float s = 0.f;
    for (int j = 0; j < KNBR; ++j) s += wgt[j];
    ssum = s;
  }
  __syncthreads();
  const float inv = 1.0f / ssum;
  const float* tb = topo + (size_t)b * SEQ * TOPO;
  float acc = 0.f;
  for (int j = 0; j < KNBR; ++j)
    acc += wgt[j] * tb[(size_t)idx[j] * TOPO + t];
  float comb = topo[(size_t)row * TOPO + t] + acc * inv;
  combH[(size_t)row * TOPO + t] = (f16)comb;
}

// ---------------------------------------------------------------------------
__global__ __launch_bounds__(256)
void layernorm_f16(const float* __restrict__ x, const float* __restrict__ scale,
                   const float* __restrict__ bias, f16* __restrict__ out)
{
  __shared__ float r1[256], r2[256];
  const int row = blockIdx.x, t = threadIdx.x;
  float v = x[(size_t)row * TOPO + t];
  r1[t] = v; r2[t] = v * v;
  __syncthreads();
  for (int s = 128; s > 0; s >>= 1) {
    if (t < s) { r1[t] += r1[t + s]; r2[t] += r2[t + s]; }
    __syncthreads();
  }
  float mu  = r1[0] * (1.0f / TOPO);
  float var = r2[0] * (1.0f / TOPO) - mu * mu;
  float nv = (v - mu) * rsqrtf(var + 1e-6f) * scale[t] + bias[t];
  out[(size_t)row * TOPO + t] = (f16)nv;
}

// ---------------------------------------------------------------------------
__global__ __launch_bounds__(256)
void prep_embed(const float* __restrict__ emb, f16* __restrict__ embH,
                f16* __restrict__ enormH)
{
  __shared__ float red[256];
  const int row = blockIdx.x, t = threadIdx.x;
  const float* e = emb + (size_t)row * EMB;
  float v0 = e[t], v1 = e[t + 256], v2 = e[t + 512];
  red[t] = v0 * v0 + v1 * v1 + v2 * v2;
  __syncthreads();
  for (int s = 128; s > 0; s >>= 1) {
    if (t < s) red[t] += red[t + s];
    __syncthreads();
  }
  float inv = 1.0f / (sqrtf(red[0]) + 1e-8f);
  f16* eh = embH   + (size_t)row * EMB;
  f16* nh = enormH + (size_t)row * EMB;
  eh[t] = (f16)v0;         eh[t + 256] = (f16)v1;         eh[t + 512] = (f16)v2;
  nh[t] = (f16)(v0 * inv); nh[t + 256] = (f16)(v1 * inv); nh[t + 512] = (f16)(v2 * inv);
}

__global__ void f32_to_f16(const float* __restrict__ s, f16* __restrict__ d, int n) {
  int i = blockIdx.x * blockDim.x + threadIdx.x;
  if (i < n) d[i] = (f16)s[i];
}

// ---------------------------------------------------------------------------
extern "C" void kernel_launch(void* const* d_in, const int* in_sizes, int n_in,
                              void* d_out, int out_size, void* d_ws, size_t ws_size,
                              hipStream_t stream)
{
  (void)in_sizes; (void)n_in; (void)out_size; (void)ws_size;
  const float* emb = (const float*)d_in[0];
  const float* w1  = (const float*)d_in[1];  const float* b1 = (const float*)d_in[2];
  const float* w2  = (const float*)d_in[3];  const float* b2 = (const float*)d_in[4];
  const float* w3  = (const float*)d_in[5];  const float* b3 = (const float*)d_in[6];
  const float* lns = (const float*)d_in[7];  const float* lnb = (const float*)d_in[8];
  const float* w4  = (const float*)d_in[9];  const float* b4 = (const float*)d_in[10];
  const float* w5  = (const float*)d_in[11]; const float* b5 = (const float*)d_in[12];

  char* ws = (char*)d_ws;
  size_t off = 0;
  auto alloc = [&](size_t bytes) -> void* {
    void* p = (void*)(ws + off);
    off += (bytes + 255) & ~(size_t)255;
    return p;
  };

  f16* w1h    = (f16*)alloc((size_t)EMB * TOPO * 2);
  f16* w2h    = (f16*)alloc((size_t)TOPO * 512 * 2);
  f16* w3h    = (f16*)alloc((size_t)512 * TOPO * 2);
  f16* w4h    = (f16*)alloc((size_t)TOPO * TOPO * 2);
  f16* w5h    = (f16*)alloc((size_t)TOPO * TOPO * 2);
  f16* embH   = (f16*)alloc((size_t)ROWS * EMB * 2);   // reused as y (f32 [ROWS,TOPO])
  f16* enormH = (f16*)alloc((size_t)ROWS * EMB * 2);   // reused as x2h (f16 [ROWS,TOPO])
  float* topo = (float*)alloc((size_t)ROWS * TOPO * 4);
  f16* combH  = (f16*)alloc((size_t)ROWS * TOPO * 2);
  f16* x1h    = (f16*)alloc((size_t)ROWS * 2 * TOPO * 2);
  f16* tfh    = (f16*)alloc((size_t)ROWS * TOPO * 2);

  float* y   = (float*)embH;   // alias: embH dead after topo GEMM
  f16*   x2h = (f16*)enormH;   // alias: enormH dead after sim kernel

  float* pers = (float*)d_out;
  float* nbrD = pers + (size_t)ROWS * TOPO;
  float* nbrI = nbrD + (size_t)ROWS * KNBR;

  f32_to_f16<<<(EMB*TOPO + 255)/256, 256, 0, stream>>>(w1, w1h, EMB*TOPO);
  f32_to_f16<<<(TOPO*512 + 255)/256, 256, 0, stream>>>(w2, w2h, TOPO*512);
  f32_to_f16<<<(512*TOPO + 255)/256, 256, 0, stream>>>(w3, w3h, 512*TOPO);
  f32_to_f16<<<(TOPO*TOPO + 255)/256, 256, 0, stream>>>(w4, w4h, TOPO*TOPO);
  f32_to_f16<<<(TOPO*TOPO + 255)/256, 256, 0, stream>>>(w5, w5h, TOPO*TOPO);

  prep_embed<<<ROWS, 256, 0, stream>>>(emb, embH, enormH);

  dim3 blk(256);
  // topo = emb @ w1 + b1                       (f32 out)
  wmma_gemm_f16<false,false><<<dim3(TOPO/64, ROWS/128), blk, 0, stream>>>(embH, w1h, b1, topo, TOPO, EMB);
  // fused similarity + top-32 -> nbr outputs
  sim_topk<<<BATCH * (SEQ/32), blk, 0, stream>>>(enormH, nbrD, nbrI);
  // combined = topo + weighted neighbors       (f16 out)
  combine_neighbors<<<ROWS, blk, 0, stream>>>(topo, nbrD, nbrI, combH);
  // x = relu(combined @ w2 + b2)               (f16 out)
  wmma_gemm_f16<true,true><<<dim3(512/64, ROWS/128), blk, 0, stream>>>(combH, w2h, b2, x1h, 512, TOPO);
  // y = x @ w3 + b3                            (f32 out)
  wmma_gemm_f16<false,false><<<dim3(TOPO/64, ROWS/128), blk, 0, stream>>>(x1h, w3h, b3, y, TOPO, 512);
  // layernorm
  layernorm_f16<<<ROWS, 256, 0, stream>>>(y, lns, lnb, tfh);
  // x = relu(tf @ w4 + b4)                     (f16 out)
  wmma_gemm_f16<true,true><<<dim3(TOPO/64, ROWS/128), blk, 0, stream>>>(tfh, w4h, b4, x2h, TOPO, TOPO);
  // persistence = x @ w5 + b5                  (f32 out)
  wmma_gemm_f16<false,false><<<dim3(TOPO/64, ROWS/128), blk, 0, stream>>>(x2h, w5h, b5, pers, TOPO, TOPO);
}
